// TripletVGfM_27015344292381
// MI455X (gfx1250) — matrix operand order repaired and checked by reading, not compiled, not run on hardware
//
#include <hip/hip_runtime.h>
#include <hip/hip_bf16.h>
#include <stdint.h>

// ---------------------------------------------------------------------------
// TripletVGfM forward for MI455X (gfx1250, wave32, WMMA + TDM)
// Heavy GEMMs (GRU gates, [M,512]x[512,1536]) run on v_wmma_f32_16x16x32_bf16
// with Tensor-Data-Mover (tensor_load_to_lds) double-buffered LDS staging.
// Gather/gate/scatter phases are wave-per-edge VALU kernels (bandwidth bound,
// served out of the 192MB L2 at 23.3 TB/s HBM behind it).
// ---------------------------------------------------------------------------

#define DEV __device__ __forceinline__

typedef __attribute__((ext_vector_type(16))) __bf16 v16bf;
typedef __attribute__((ext_vector_type(8)))  float  v8f;

#if __has_builtin(__builtin_amdgcn_tensor_load_to_lds) && \
    __has_builtin(__builtin_amdgcn_s_wait_tensorcnt)
#define USE_TDM 1
typedef unsigned int u32x4 __attribute__((ext_vector_type(4)));
typedef int          i32x8 __attribute__((ext_vector_type(8)));
typedef int          i32x4 __attribute__((ext_vector_type(4)));
#endif

union FragU {
    v16bf v;
    unsigned int u[8];
};

DEV float sigmoidf_(float x) { return 1.0f / (1.0f + __expf(-x)); }

DEV unsigned short f2bf(float f) {
    union { float f; unsigned int u; } v; v.f = f;
    unsigned int r = v.u + 0x7FFFu + ((v.u >> 16) & 1u);  // round-to-nearest-even
    return (unsigned short)(r >> 16);
}

// ---------------------------------------------------------------------------
// Elementwise utility kernels
// ---------------------------------------------------------------------------
__global__ void fill_f32(float* __restrict__ p, float v, long n) {
    long i = (long)blockIdx.x * blockDim.x + threadIdx.x;
    if (i < n) p[i] = v;
}

__global__ void cvt_f32_bf16(const float* __restrict__ in,
                             unsigned short* __restrict__ out, long n) {
    long i = (long)blockIdx.x * blockDim.x + threadIdx.x;
    if (i < n) out[i] = f2bf(in[i]);
}

// geo encoder: relu(geo_feature @ W^T + b), W:[512,11]
__global__ void geo_encode(const float* __restrict__ gf, const float* __restrict__ W,
                           const float* __restrict__ b, float* __restrict__ out, int E) {
    long i = (long)blockIdx.x * blockDim.x + threadIdx.x;
    if (i >= (long)E * 512) return;
    int e = (int)(i >> 9);
    int d = (int)(i & 511);
    float s = b[d];
    #pragma unroll
    for (int g = 0; g < 11; ++g) s += gf[(long)e * 11 + g] * W[d * 11 + g];
    out[i] = s > 0.f ? s : 0.f;
}

#if USE_TDM
// Build TDM descriptor group1 for a 2D bf16 tensor [rows, 512], tile 128x32.
// (ISA 8.4: data_size code 1 = 2 bytes; tile_dim0=32 (X, contiguous),
//  tile_dim1=128 rows; tensor_dim0=512; tensor_dim0_stride=512.)
DEV i32x8 tdm_group1(unsigned rows) {
    i32x8 g;
    g[0] = (int)(1u << 16);                               // data_size=1 (2B)
    g[1] = (int)((512u & 0xFFFFu) << 16);                 // tensor_dim0[15:0]
    g[2] = (int)((512u >> 16) | ((rows & 0xFFFFu) << 16));// dim0[31:16] | dim1[15:0]
    g[3] = (int)((rows >> 16) | (32u << 16));             // dim1[31:16] | tile_dim0=32
    g[4] = (int)(128u);                                   // tile_dim1=128, tile_dim2=0
    g[5] = (int)(512u);                                   // tensor_dim0_stride[31:0]
    g[6] = 0;                                             // stride[47:32] | dim1_stride lo
    g[7] = 0;
    return g;
}

// Issue one TDM tile load: global (byte addr) -> LDS (byte addr).
DEV void tdm_issue(const void* gaddr, unsigned lds_addr, const i32x8& g1) {
    unsigned long long ga = (unsigned long long)(uintptr_t)gaddr;
    u32x4 g0;
    g0[0] = 1u;                                            // count=1 user descriptor
    g0[1] = lds_addr;                                      // lds_addr (bytes)
    g0[2] = (unsigned)(ga & 0xFFFFFFFFull);                // global_addr[31:0]
    g0[3] = (unsigned)((ga >> 32) & 0x1FFFFFFull) | (2u << 30); // addr[56:32]|type=2
    i32x4 z4 = {0, 0, 0, 0};                               // groups 2/3 unused (2D)
    i32x8 z8 = {0, 0, 0, 0, 0, 0, 0, 0};
    // clang-23 / therock-10.0 6-arg form:
    // (g0 u32x4, g1 i32x8, g2 i32x4, g3 i32x4, extra i32x8, cpol i32)
    __builtin_amdgcn_tensor_load_to_lds(g0, g1, z4, z4, z8, 0);
}
#endif

// ---------------------------------------------------------------------------
// WMMA GEMM: C[M,1536] = A[M,512](bf16) @ W[1536,512](bf16)^T + bias[1536]
// Block = 256 threads (8 waves). Block tile 128x128, K-slab 32, double-
// buffered LDS filled by the Tensor Data Mover (wave 0 issues, TENSORcnt
// waits, split-barrier publishes). Wave tile 64x32 -> 4x2 16x16 accumulators.
// ---------------------------------------------------------------------------
__global__ void __launch_bounds__(256)
gemm_bf16_wmma(const unsigned short* __restrict__ A,
               const unsigned short* __restrict__ W,
               const float* __restrict__ bias,
               float* __restrict__ C, int M) {
    constexpr int K = 512;
    constexpr int NT = 1536;
    constexpr int NKT = K / 32;                 // 16 k-tiles
    __shared__ __align__(16) unsigned short lA[2][128 * 32];
    __shared__ __align__(16) unsigned short lB[2][128 * 32];

    const int tid  = threadIdx.x;
    const int lane = tid & 31;
    const int wave = tid >> 5;
    const int m0   = blockIdx.y * 128;
    const int n0   = blockIdx.x * 128;
    const int wm0  = (wave & 1) * 64;   // 2 waves down
    const int wn0  = (wave >> 1) * 32;  // 4 waves across

    v8f acc[4][2];
    #pragma unroll
    for (int i = 0; i < 4; ++i)
        #pragma unroll
        for (int j = 0; j < 2; ++j)
            #pragma unroll
            for (int r = 0; r < 8; ++r) acc[i][j][r] = 0.0f;

    // CDNA5 WMMA 16-bit operand lane layouts (ISA 7.12.2)
    const int kbA = (lane & 16) ? 8 : 0;    // A: lanes 16-31 hold K=8..15 / 24..31
    const int mA  = lane & 15;
    const int kbB = (lane & 16) ? 8 : 0;    // B: lanes 16-31 hold K=8..15 / 24..31
    const int nB  = lane & 15;

#if USE_TDM
    const i32x8 g1A = tdm_group1((unsigned)M);
    const i32x8 g1B = tdm_group1((unsigned)NT);
    const unsigned ldsA0 = (unsigned)(uintptr_t)&lA[0][0];
    const unsigned ldsB0 = (unsigned)(uintptr_t)&lB[0][0];
    constexpr unsigned TILE_BYTES = 128 * 32 * 2;

    if (wave == 0) {  // prologue: tile 0 -> buffer 0
        tdm_issue(A + (size_t)m0 * K, ldsA0, g1A);
        tdm_issue(W + (size_t)n0 * K, ldsB0, g1B);
    }
    for (int kt = 0; kt < NKT; ++kt) {
        const int buf = kt & 1;
        if (wave == 0) {
            if (kt + 1 < NKT) {  // issue next tile into the other buffer
                int k1 = (kt + 1) * 32;
                tdm_issue(A + (size_t)m0 * K + k1, ldsA0 + (buf ^ 1) * TILE_BYTES, g1A);
                tdm_issue(W + (size_t)n0 * K + k1, ldsB0 + (buf ^ 1) * TILE_BYTES, g1B);
                __builtin_amdgcn_s_wait_tensorcnt(2);  // tile kt finished
            } else {
                __builtin_amdgcn_s_wait_tensorcnt(0);
            }
        }
        __syncthreads();  // publish buf to all waves
        const unsigned short* baseA = &lA[buf][0];
        const unsigned short* baseB = &lB[buf][0];
#else
    const int ldRow  = tid >> 2;   // 0..63
    const int ldQuad = tid & 3;    // 0..3  (8 bf16 = 16B each)
    for (int kt = 0; kt < NKT; ++kt) {
        const int k0 = kt * 32;
        __syncthreads();
        #pragma unroll
        for (int half = 0; half < 2; ++half) {
            int r = ldRow + half * 64;
            *(uint4*)(&lA[0][r * 32 + ldQuad * 8]) =
                *(const uint4*)(&A[(size_t)(m0 + r) * K + k0 + ldQuad * 8]);
            *(uint4*)(&lB[0][r * 32 + ldQuad * 8]) =
                *(const uint4*)(&W[(size_t)(n0 + r) * K + k0 + ldQuad * 8]);
        }
        if (k0 + 32 < K) {
            __builtin_prefetch(&A[(size_t)(m0 + ldRow) * K + k0 + 32], 0, 3);
            __builtin_prefetch(&W[(size_t)(n0 + ldRow) * K + k0 + 32], 0, 3);
        }
        __syncthreads();
        const unsigned short* baseA = &lA[0][0];
        const unsigned short* baseB = &lB[0][0];
#endif
        FragU bf[2];
        #pragma unroll
        for (int j = 0; j < 2; ++j) {
            const unsigned short* p = &baseB[(wn0 + j * 16 + nB) * 32 + kbB];
            *(uint4*)(&bf[j].u[0]) = *(const uint4*)(p);
            *(uint4*)(&bf[j].u[4]) = *(const uint4*)(p + 16);
        }
        #pragma unroll
        for (int i = 0; i < 4; ++i) {
            FragU af;
            const unsigned short* p = &baseA[(wm0 + i * 16 + mA) * 32 + kbA];
            *(uint4*)(&af.u[0]) = *(const uint4*)(p);        // K = kb .. kb+7
            *(uint4*)(&af.u[4]) = *(const uint4*)(p + 16);   // K = 16+kb .. 16+kb+7
            #pragma unroll
            for (int j = 0; j < 2; ++j) {
                acc[i][j] = __builtin_amdgcn_wmma_f32_16x16x32_bf16(
                    false, af.v, false, bf[j].v, (short)0, acc[i][j], false, false);
            }
        }
#if USE_TDM
        __syncthreads();  // all reads of buf done before TDM refills it
#endif
    }

    const int rowAdd = (lane & 16) ? 8 : 0;  // C/D layout: lanes 16-31 -> M+8
    #pragma unroll
    for (int i = 0; i < 4; ++i) {
        #pragma unroll
        for (int j = 0; j < 2; ++j) {
            int col = n0 + wn0 + j * 16 + nB;
            float bv = bias[col];
            int rbase = m0 + wm0 + i * 16 + rowAdd;
            #pragma unroll
            for (int r = 0; r < 8; ++r)
                C[(size_t)(rbase + r) * NT + col] = acc[i][j][r] + bv;
        }
    }
}

// ---------------------------------------------------------------------------
// GRU nonlinearity: h' = (1-z)*n + z*h  with  r,z,n from gi (GI) and gh (GH).
// gh_is_bias: GH is just bhh[1536] (hidden state == 0 at init).
// ---------------------------------------------------------------------------
__global__ void gru_combine(const float* __restrict__ GI, const float* __restrict__ GH,
                            int gh_is_bias, const float* __restrict__ Hin,
                            float* __restrict__ Hout, int M) {
    long i = (long)blockIdx.x * blockDim.x + threadIdx.x;
    if (i >= (long)M * 512) return;
    int m = (int)(i >> 9);
    int c = (int)(i & 511);
    long base = (long)m * 1536;
    float ir = GI[base + c], iz = GI[base + 512 + c], in_ = GI[base + 1024 + c];
    float hr, hz, hn;
    if (gh_is_bias) { hr = GH[c]; hz = GH[512 + c]; hn = GH[1024 + c]; }
    else { hr = GH[base + c]; hz = GH[base + 512 + c]; hn = GH[base + 1024 + c]; }
    float r  = sigmoidf_(ir + hr);
    float z  = sigmoidf_(iz + hz);
    float nn = tanhf(in_ + r * hn);
    float h  = (Hin != nullptr) ? Hin[i] : 0.0f;
    Hout[i] = (1.0f - z) * nn + z * h;
}

// ---------------------------------------------------------------------------
// Main edge phase: one wave (32 lanes) per edge computes 5 gates (10 dots of
// length 512), scatters node messages (atomics) and writes edge messages.
// ---------------------------------------------------------------------------
__global__ void __launch_bounds__(256)
edge_gates(const float* __restrict__ x, const float* __restrict__ ef,
           const float* __restrict__ geo, const int* __restrict__ eidx,
           const float* __restrict__ Wsn, const float* __restrict__ bsn,
           const float* __restrict__ Won, const float* __restrict__ bon,
           const float* __restrict__ Wse, const float* __restrict__ bse,
           const float* __restrict__ Woe, const float* __restrict__ boe,
           const float* __restrict__ Wge, const float* __restrict__ bge,
           float* __restrict__ node_sum, float* __restrict__ cnt_node,
           float* __restrict__ edge_msg, int E) {
    int wid  = (int)(((long)blockIdx.x * blockDim.x + threadIdx.x) >> 5);
    int lane = threadIdx.x & 31;
    if (wid >= E) return;
    int s  = eidx[wid];       // src -> x_j
    int dn = eidx[E + wid];   // dst -> x_i
    const float* xi = x + (long)dn * 512;
    const float* xj = x + (long)s * 512;
    const float* ev = ef + (long)wid * 512;
    const float* gv = geo + (long)wid * 512;

    float acc[10];
    #pragma unroll
    for (int t = 0; t < 10; ++t) acc[t] = 0.0f;
    for (int d = lane; d < 512; d += 32) {
        float vi = xi[d], vj = xj[d], ve = ev[d], vg = gv[d];
        acc[0] += vi * Wsn[d];  acc[1] += ve * Wsn[512 + d];
        acc[2] += vj * Won[d];  acc[3] += ve * Won[512 + d];
        acc[4] += vi * Wse[d];  acc[5] += ve * Wse[512 + d];
        acc[6] += vj * Woe[d];  acc[7] += ve * Woe[512 + d];
        acc[8] += vg * Wge[d];  acc[9] += ve * Wge[512 + d];
    }
    #pragma unroll
    for (int t = 0; t < 10; ++t)
        #pragma unroll
        for (int off = 16; off > 0; off >>= 1)
            acc[t] += __shfl_xor(acc[t], off, 32);

    float gsn = sigmoidf_(acc[0] + acc[1] + bsn[0]);
    float gon = sigmoidf_(acc[2] + acc[3] + bon[0]);
    float gse = sigmoidf_(acc[4] + acc[5] + bse[0]);
    float goe = sigmoidf_(acc[6] + acc[7] + boe[0]);
    float gge = sigmoidf_(acc[8] + acc[9] + bge[0]);
    float gn  = gsn + gon;

    for (int d = lane; d < 512; d += 32) {
        atomicAdd(&node_sum[(long)dn * 512 + d], gn * ev[d]);
        edge_msg[(long)wid * 512 + d] = gse * xi[d] + goe * xj[d] + gge * gv[d];
    }
    if (lane == 0) atomicAdd(&cnt_node[dn], 1.0f);
}

// Temporal gate message: g = sigmoid(f[dst]·Wa + f[src]·Wb + b); sum[dst]+=g*f[dst]
__global__ void __launch_bounds__(256)
temporal_gate(const float* __restrict__ feat, const int* __restrict__ idx, int M,
              const float* __restrict__ W, const float* __restrict__ b,
              float* __restrict__ sum, float* __restrict__ cnt) {
    int wid  = (int)(((long)blockIdx.x * blockDim.x + threadIdx.x) >> 5);
    int lane = threadIdx.x & 31;
    if (wid >= M) return;
    int s  = idx[wid];
    int dn = idx[M + wid];
    const float* fi = feat + (long)dn * 512;
    const float* fj = feat + (long)s * 512;
    float a0 = 0.f, a1 = 0.f;
    for (int d = lane; d < 512; d += 32) { a0 += fi[d] * W[d]; a1 += fj[d] * W[512 + d]; }
    #pragma unroll
    for (int off = 16; off > 0; off >>= 1) {
        a0 += __shfl_xor(a0, off, 32);
        a1 += __shfl_xor(a1, off, 32);
    }
    float g = sigmoidf_(a0 + a1 + b[0]);
    for (int d = lane; d < 512; d += 32)
        atomicAdd(&sum[(long)dn * 512 + d], g * fi[d]);
    if (lane == 0) atomicAdd(&cnt[dn], 1.0f);
}

__global__ void finalize_node(float* __restrict__ node_msg, const float* __restrict__ cnt_n,
                              const float* __restrict__ tn_sum, const float* __restrict__ cnt_tn,
                              int N) {
    long i = (long)blockIdx.x * blockDim.x + threadIdx.x;
    if (i >= (long)N * 512) return;
    int n = (int)(i >> 9);
    float c1 = cnt_n[n];  c1 = c1 > 1.f ? c1 : 1.f;
    float c2 = cnt_tn[n]; c2 = c2 > 1.f ? c2 : 1.f;
    node_msg[i] = node_msg[i] / c1 + tn_sum[i] / c2;
}

__global__ void finalize_edge(float* __restrict__ edge_msg, const float* __restrict__ te_sum,
                              const float* __restrict__ cnt_te, int E) {
    long i = (long)blockIdx.x * blockDim.x + threadIdx.x;
    if (i >= (long)E * 512) return;
    int e = (int)(i >> 9);
    float c = cnt_te[e]; c = c > 1.f ? c : 1.f;
    edge_msg[i] += te_sum[i] / c;
}

// ---------------------------------------------------------------------------
// Host orchestration
// ---------------------------------------------------------------------------
extern "C" void kernel_launch(void* const* d_in, const int* in_sizes, int n_in,
                              void* d_out, int out_size, void* d_ws, size_t ws_size,
                              hipStream_t stream) {
    (void)in_sizes; (void)n_in; (void)out_size; (void)ws_size;
    const int N = 8192, E = 65536, D = 512, TNc = 16384, TEc = 32768;
    const int TD = 1536;           // 3*D
    const int CH = 8192;           // GEMM row chunk
    const int NCHUNK = E / CH;     // 8

    const float* x_in  = (const float*)d_in[0];
    const float* ef_in = (const float*)d_in[1];
    const float* geof  = (const float*)d_in[2];
    const int*   eidx  = (const int*)d_in[3];
    const int*   tng   = (const int*)d_in[4];
    const int*   teg   = (const int*)d_in[5];
    const float* nWih = (const float*)d_in[6];
    const float* nWhh = (const float*)d_in[7];
    const float* nbih = (const float*)d_in[8];
    const float* nbhh = (const float*)d_in[9];
    const float* eWih = (const float*)d_in[10];
    const float* eWhh = (const float*)d_in[11];
    const float* ebih = (const float*)d_in[12];
    const float* ebhh = (const float*)d_in[13];
    const float* gW[7]; const float* gb[7];
    for (int i = 0; i < 7; ++i) {
        gW[i] = (const float*)d_in[14 + 2 * i];
        gb[i] = (const float*)d_in[15 + 2 * i];
    }
    const float* geoW = (const float*)d_in[28];
    const float* geob = (const float*)d_in[29];

    // ---- workspace carve (256B aligned slices) ----
    char* wp = (char*)d_ws;
    auto carve = [&](size_t bytes) {
        char* r = wp;
        wp += (bytes + 255) & ~(size_t)255;
        return r;
    };
    float* h_node   = (float*)carve((size_t)N * D * 4);
    float* h_edge   = (float*)carve((size_t)E * D * 4);
    float* geo      = (float*)carve((size_t)E * D * 4);
    float* node_msg = (float*)carve((size_t)N * D * 4);   // also node scatter-sum
    float* edge_msg = (float*)carve((size_t)E * D * 4);
    // union region: {te_sum[E*D] + tn_sum[N*D]} during gate phase,
    //               {GI[CH*TD] + GH[CH*TD]} during GEMM phase (disjoint in time)
    size_t uBytes = (size_t)E * D * 4 + (size_t)N * D * 4;
    {
        size_t gBytes = 2 * (size_t)CH * TD * 4;
        if (gBytes > uBytes) uBytes = gBytes;
    }
    char* U = carve(uBytes);
    float* te_sum = (float*)U;
    float* tn_sum = (float*)(U + (size_t)E * D * 4);
    float* GI     = (float*)U;
    float* GH     = (float*)(U + (size_t)CH * TD * 4);
    float* cnt_node = (float*)carve((size_t)N * 4);
    float* cnt_tn   = (float*)carve((size_t)N * 4);
    float* cnt_te   = (float*)carve((size_t)E * 4);
    unsigned short* bWihn = (unsigned short*)carve((size_t)TD * D * 2);
    unsigned short* bWhhn = (unsigned short*)carve((size_t)TD * D * 2);
    unsigned short* bWihe = (unsigned short*)carve((size_t)TD * D * 2);
    unsigned short* bWhhe = (unsigned short*)carve((size_t)TD * D * 2);
    unsigned short* Abf = (unsigned short*)carve((size_t)CH * D * 2);
    unsigned short* Hbf = (unsigned short*)carve((size_t)CH * D * 2);

    auto g1 = [](long n) { return dim3((unsigned)((n + 255) / 256)); };
    auto cvt = [&](const float* src, unsigned short* dst, long n) {
        cvt_f32_bf16<<<g1(n), 256, 0, stream>>>(src, dst, n);
    };
    auto gemm = [&](const unsigned short* A, const unsigned short* W,
                    const float* bias, float* C, int M) {
        dim3 grid(TD / 128, M / 128);
        gemm_bf16_wmma<<<grid, 256, 0, stream>>>(A, W, bias, C, M);
    };
    auto zero = [&](float* p, long n) {
        fill_f32<<<g1(n), 256, 0, stream>>>(p, 0.0f, n);
    };

    // ---- one-time prep ----
    cvt(nWih, bWihn, (long)TD * D);
    cvt(nWhh, bWhhn, (long)TD * D);
    cvt(eWih, bWihe, (long)TD * D);
    cvt(eWhh, bWhhe, (long)TD * D);
    geo_encode<<<g1((long)E * D), 256, 0, stream>>>(geof, geoW, geob, geo, E);

    // ---- initial GRU (hx = 0); node_gru used for BOTH x and edge_feature ----
    cvt(x_in, Abf, (long)N * D);
    gemm(Abf, bWihn, nbih, GI, N);
    gru_combine<<<g1((long)N * D), 256, 0, stream>>>(GI, nbhh, 1, nullptr, h_node, N);
    for (int c = 0; c < NCHUNK; ++c) {
        cvt(ef_in + (size_t)c * CH * D, Abf, (long)CH * D);
        gemm(Abf, bWihn, nbih, GI, CH);
        gru_combine<<<g1((long)CH * D), 256, 0, stream>>>(
            GI, nbhh, 1, nullptr, h_edge + (size_t)c * CH * D, CH);
    }

    // ---- message-passing layers ----
    for (int layer = 0; layer < 2; ++layer) {
        zero(node_msg, (long)N * D);
        zero(tn_sum, (long)N * D);
        zero(te_sum, (long)E * D);
        zero(cnt_node, N);
        zero(cnt_tn, N);
        zero(cnt_te, E);

        edge_gates<<<dim3(E / 8), 256, 0, stream>>>(
            h_node, h_edge, geo, eidx,
            gW[0], gb[0], gW[1], gb[1], gW[2], gb[2], gW[3], gb[3], gW[4], gb[4],
            node_msg, cnt_node, edge_msg, E);
        temporal_gate<<<dim3(TNc / 8), 256, 0, stream>>>(
            h_node, tng, TNc, gW[5], gb[5], tn_sum, cnt_tn);
        temporal_gate<<<dim3(TEc / 8), 256, 0, stream>>>(
            h_edge, teg, TEc, gW[6], gb[6], te_sum, cnt_te);

        finalize_node<<<g1((long)N * D), 256, 0, stream>>>(node_msg, cnt_node, tn_sum, cnt_tn, N);
        finalize_edge<<<g1((long)E * D), 256, 0, stream>>>(edge_msg, te_sum, cnt_te, E);

        // node GRU
        cvt(node_msg, Abf, (long)N * D);
        cvt(h_node, Hbf, (long)N * D);
        gemm(Abf, bWihn, nbih, GI, N);
        gemm(Hbf, bWhhn, nbhh, GH, N);
        gru_combine<<<g1((long)N * D), 256, 0, stream>>>(GI, GH, 0, h_node, h_node, N);

        // edge GRU (chunked to bound GI/GH footprint)
        for (int c = 0; c < NCHUNK; ++c) {
            float* hptr = h_edge + (size_t)c * CH * D;
            cvt(edge_msg + (size_t)c * CH * D, Abf, (long)CH * D);
            cvt(hptr, Hbf, (long)CH * D);
            gemm(Abf, bWihe, ebih, GI, CH);
            gemm(Hbf, bWhhe, ebhh, GH, CH);
            gru_combine<<<g1((long)CH * D), 256, 0, stream>>>(GI, GH, 0, hptr, hptr, CH);
        }
    }

    // ---- outputs: concat(x, edge_feature) ----
    (void)hipMemcpyAsync(d_out, h_node, (size_t)N * D * 4, hipMemcpyDeviceToDevice, stream);
    (void)hipMemcpyAsync((float*)d_out + (size_t)N * D, h_edge, (size_t)E * D * 4,
                         hipMemcpyDeviceToDevice, stream);
}
